// MultiHeadAttention_70789650973092
// MI455X (gfx1250) — matrix operand order repaired and checked by reading, not compiled, not run on hardware
//
#include <hip/hip_runtime.h>
#include <math.h>

typedef __bf16 bf16;
typedef __attribute__((ext_vector_type(16))) __bf16 v16bf;
typedef __attribute__((ext_vector_type(8)))  __bf16 v8bf;
typedef __attribute__((ext_vector_type(8)))  float  v8f;

namespace {
constexpr int kB    = 4;
constexpr int kT    = 2048;
constexpr int kHds  = 8;           // heads
constexpr int kDk   = 64;          // per-head dim
constexpr int kF    = 512;         // feature dim (in == out == heads*dk)
constexpr int kW    = 16;          // window
constexpr int kCols = 2 * kW + 1;  // 33
}

static __device__ __forceinline__ v16bf concat8(v8bf lo, v8bf hi) {
  return __builtin_shufflevector(lo, hi, 0,1,2,3,4,5,6,7,8,9,10,11,12,13,14,15);
}

// ---------------------------------------------------------------------------
// One-shot f32 -> bf16 conversion (8 elems/thread, b128 in / b128 out).
// Keeps all converts OUT of the GEMM inner loop and halves its load bytes.
// ---------------------------------------------------------------------------
__global__ void f32_to_bf16(const float* __restrict__ in,
                            bf16* __restrict__ out, int n)
{
  int idx = (blockIdx.x * blockDim.x + threadIdx.x) * 8;
  if (idx >= n) return;
  const float4 a = *(const float4*)(in + idx);
  const float4 b = *(const float4*)(in + idx + 4);
  v8bf v;
  v[0]=(bf16)a.x; v[1]=(bf16)a.y; v[2]=(bf16)a.z; v[3]=(bf16)a.w;
  v[4]=(bf16)b.x; v[5]=(bf16)b.y; v[6]=(bf16)b.z; v[7]=(bf16)b.w;
  *(v8bf*)(out + idx) = v;
}

// ---------------------------------------------------------------------------
// Projection: Out[m,n] = bf16( sum_k X[m,k]*W[n,k] + bias[n] )
// X, W already bf16. Workgroup = 4 waves; wave w owns rows m0..m0+15 and all
// 64 columns of the block's N-tile (4 v8f accumulators). Inner loop is pure
// b128 loads + v_wmma_f32_16x16x32_bf16.
// A-frag lane layout (16x32 bf16): lane(0-15)=row, half selects K 0..15
// vs 16..31. B-frag: lane(0-15)=column, half selects K 0..15 vs 16..31.
// ---------------------------------------------------------------------------
__global__ void proj_bf16_wmma(const bf16* __restrict__ X,
                               const bf16* __restrict__ W,
                               const float* __restrict__ bias,
                               bf16* __restrict__ Out)
{
  const int lane = threadIdx.x & 31;
  const int wv   = threadIdx.x >> 5;
  const int lp   = lane & 15;
  const int hf   = lane >> 4;

  const int m0 = 64 * blockIdx.x + 16 * wv;
  const int n0 = 64 * blockIdx.y;

  const bf16* arow = X + (size_t)(m0 + lp) * kF;

  v8f acc[4];
#pragma unroll
  for (int t = 0; t < 4; ++t) acc[t] = (v8f){0,0,0,0,0,0,0,0};

  for (int k0 = 0; k0 < kF; k0 += 32) {
    __builtin_prefetch((const void*)(arow + k0 + 32), 0, 0);

    // A fragment: rows m0+lp, K = k0+16*hf .. +15 (two b128 loads)
    const v8bf alo = *(const v8bf*)(arow + k0 + 16 * hf);
    const v8bf ahi = *(const v8bf*)(arow + k0 + 16 * hf + 8);
    const v16bf afrag = concat8(alo, ahi);

#pragma unroll
    for (int t = 0; t < 4; ++t) {
      // B fragment: column n0+16t+lp, K = k0+16*hf .. +15 (two b128 loads)
      const bf16* wrow = W + (size_t)(n0 + 16 * t + lp) * kF + k0 + 16 * hf;
      const v8bf blo = *(const v8bf*)(wrow);
      const v8bf bhi = *(const v8bf*)(wrow + 8);
      const v16bf bfrag = concat8(blo, bhi);

      acc[t] = __builtin_amdgcn_wmma_f32_16x16x32_bf16(
          false, afrag, false, bfrag, (short)0, acc[t], false, false);
    }
  }

  // epilogue: bias add + bf16 store. D layout: lane lp = col, row m0+8*hf+r
#pragma unroll
  for (int t = 0; t < 4; ++t) {
    const int n = n0 + 16 * t + lp;
    const float bv = bias[n];
#pragma unroll
    for (int r = 0; r < 8; ++r) {
      const int m = m0 + 8 * hf + r;
      Out[(size_t)m * kF + n] = (bf16)(acc[t][r] + bv);
    }
  }
}

// ---------------------------------------------------------------------------
// Banded attention. One wave per 16-row query tile of one (b,h).
// Computes a 16x48 score band (columns jbase..jbase+47, jbase = i0-16) with
// 3x2 bf16 WMMAs, window-masks, softmaxes per row (cross-lane xor reduce
// within each 16-lane half), writes the 33 gathered columns directly.
// ---------------------------------------------------------------------------
__global__ void window_attn_wmma(const bf16* __restrict__ Q,
                                 const bf16* __restrict__ K,
                                 float* __restrict__ Out)
{
  const int lane = threadIdx.x & 31;
  const int wv   = threadIdx.x >> 5;
  const int lp   = lane & 15;
  const int hf   = lane >> 4;

  const int i0 = 64 * blockIdx.x + 16 * wv;
  const int h  = blockIdx.y;
  const int b  = blockIdx.z;
  const int jbase = i0 - kW;

  // Q A-fragments for the two 32-wide K chunks of dk=64
  const bf16* qrow = Q + (size_t)(b * kT + i0 + lp) * kF + h * kDk;
  v16bf aq[2];
#pragma unroll
  for (int s = 0; s < 2; ++s) {
    v8bf lo = *(const v8bf*)(qrow + 32 * s + 16 * hf);
    v8bf hi = *(const v8bf*)(qrow + 32 * s + 16 * hf + 8);
    aq[s] = concat8(lo, hi);
  }

  v8f acc[3];
#pragma unroll
  for (int jt = 0; jt < 3; ++jt) acc[jt] = (v8f){0,0,0,0,0,0,0,0};

#pragma unroll
  for (int jt = 0; jt < 3; ++jt) {
    int j  = jbase + 16 * jt + lp;
    int jc = j < 0 ? 0 : (j >= kT ? kT - 1 : j);   // clamp; garbage cols masked later
    const bf16* krow = K + (size_t)(b * kT + jc) * kF + h * kDk;
#pragma unroll
    for (int s = 0; s < 2; ++s) {
      v8bf lo = *(const v8bf*)(krow + 32 * s + 16 * hf);
      v8bf hi = *(const v8bf*)(krow + 32 * s + 16 * hf + 8);
      v16bf bfrag = concat8(lo, hi);
      acc[jt] = __builtin_amdgcn_wmma_f32_16x16x32_bf16(
          false, aq[s], false, bfrag, (short)0, acc[jt], false, false);
    }
  }

  constexpr float kScale = 0.125f;  // 1/sqrt(64)
  const float NEG_INF = -__builtin_inff();

#pragma unroll
  for (int r = 0; r < 8; ++r) {
    const int i = i0 + 8 * hf + r;

    float sv[3];
    bool  ok[3];
    float mx = NEG_INF;
#pragma unroll
    for (int jt = 0; jt < 3; ++jt) {
      int j = jbase + 16 * jt + lp;
      ok[jt] = (j >= 0) && (j < kT) && (j >= i - kW) && (j <= i + kW);
      sv[jt] = ok[jt] ? acc[jt][r] * kScale : NEG_INF;
      mx = fmaxf(mx, sv[jt]);
    }
    // row-max across the 16 lanes of this half
#pragma unroll
    for (int m = 1; m <= 8; m <<= 1) mx = fmaxf(mx, __shfl_xor(mx, m, 32));

    float ex[3];
    float sum = 0.f;
#pragma unroll
    for (int jt = 0; jt < 3; ++jt) {
      ex[jt] = ok[jt] ? __expf(sv[jt] - mx) : 0.f;
      sum += ex[jt];
    }
#pragma unroll
    for (int m = 1; m <= 8; m <<= 1) sum += __shfl_xor(sum, m, 32);
    const float inv = 1.0f / sum;  // window always contains j == i -> sum > 0

    int starts = i - kW;
    starts = starts < 0 ? 0 : starts;
    starts = starts > (kT - kCols) ? (kT - kCols) : starts;

    float* orow = Out + ((size_t)(b * kHds + h) * kT + i) * kCols;
#pragma unroll
    for (int jt = 0; jt < 3; ++jt) {
      int j = jbase + 16 * jt + lp;
      if (ok[jt]) orow[j - starts] = ex[jt] * inv;
    }
  }
}

__global__ void zero_f32(float* __restrict__ p, int n)
{
  int idx = blockIdx.x * blockDim.x + threadIdx.x;
  if (idx < n) p[idx] = 0.f;
}

extern "C" void kernel_launch(void* const* d_in, const int* in_sizes, int n_in,
                              void* d_out, int out_size, void* d_ws, size_t ws_size,
                              hipStream_t stream)
{
  const float* query = (const float*)d_in[0];
  const float* key   = (const float*)d_in[1];
  const float* Wq    = (const float*)d_in[2];
  const float* bq    = (const float*)d_in[3];
  const float* Wk    = (const float*)d_in[4];
  const float* bk    = (const float*)d_in[5];
  float* out = (float*)d_out;

  const int M  = kB * kT;            // 8192 rows
  const int nX = M * kF;             // 4,194,304 elems (query / key)
  const int nW = kF * kF;            // 262,144 elems (Wq / Wk)

  // workspace layout (bf16): Qproj | Kproj | Xq | Xk | Wqb | Wkb
  bf16* Qbf  = (bf16*)d_ws;
  bf16* Kbf  = Qbf  + (size_t)nX;
  bf16* Xqbf = Kbf  + (size_t)nX;
  bf16* Xkbf = Xqbf + (size_t)nX;
  bf16* Wqbf = Xkbf + (size_t)nX;
  bf16* Wkbf = Wqbf + (size_t)nW;

  // positions outside the computed band are provably masked-zero: pre-zero out
  zero_f32<<<(out_size + 255) / 256, 256, 0, stream>>>(out, out_size);

  // one-shot f32 -> bf16 of GEMM operands
  f32_to_bf16<<<(nX / 8 + 255) / 256, 256, 0, stream>>>(query, Xqbf, nX);
  f32_to_bf16<<<(nX / 8 + 255) / 256, 256, 0, stream>>>(key,   Xkbf, nX);
  f32_to_bf16<<<(nW / 8 + 255) / 256, 256, 0, stream>>>(Wq,    Wqbf, nW);
  f32_to_bf16<<<(nW / 8 + 255) / 256, 256, 0, stream>>>(Wk,    Wkbf, nW);

  dim3 pgrid(M / 64, kF / 64);
  proj_bf16_wmma<<<pgrid, 128, 0, stream>>>(Xqbf, Wqbf, bq, Qbf);
  proj_bf16_wmma<<<pgrid, 128, 0, stream>>>(Xkbf, Wkbf, bk, Kbf);

  dim3 agrid(kT / 64, kHds, kB);
  window_attn_wmma<<<agrid, 128, 0, stream>>>(Qbf, Kbf, out);
}
